// PolarRnn_34179349741661
// MI455X (gfx1250) — matrix operand order repaired and checked by reading, not compiled, not run on hardware
//
#include <hip/hip_runtime.h>
#include <hip/hip_bf16.h>
#include <math.h>

typedef __bf16 bf16;
typedef bf16 v16bf __attribute__((ext_vector_type(16)));
typedef bf16 v8bf  __attribute__((ext_vector_type(8)));
typedef float v8f  __attribute__((ext_vector_type(8)));

#define B_  4
#define N_  4096
#define D_  1024
#define H_  16
#define HD_ 64
#define M_  (B_ * N_)            // 16384 tokens
#define NC_ (3 * D_ + H_ + HD_)  // 3152 fused projection columns

// ---------------------------------------------------------------------------
// WMMA helpers (CDNA5, wave32): D = A(16x32 bf16) x B(32x16 bf16) + C(f32)
// ---------------------------------------------------------------------------
static __device__ __forceinline__ v8f wmma_bf(v16bf a, v16bf b, v8f c) {
  return __builtin_amdgcn_wmma_f32_16x16x32_bf16(false, a, false, b, (short)0, c,
                                                 false, false);
}

// A fragment from row-major [m][k] LDS tile. Lane l<16 holds row m=l,
// elems 0..7 -> k=0..7, 8..15 -> k=16..23; lanes 16..31 hold k+8.
static __device__ __forceinline__ v16bf load_a_frag(const bf16* base, int stride,
                                                    int lane) {
  int mr = lane & 15, kh = lane >> 4;
  const bf16* p = base + mr * stride + kh * 8;
  v8bf lo = *(const v8bf*)p;
  v8bf hi = *(const v8bf*)(p + 16);
  v16bf a;
#pragma unroll
  for (int i = 0; i < 8; ++i) { a[i] = lo[i]; a[i + 8] = hi[i]; }
  return a;
}

// B fragment from n-major [n][k] LDS tile. Lane l holds column n=l%16,
// elems 0..7 -> k=(l/16)*8+0..7, 8..15 -> k=16+(l/16)*8+0..7.
static __device__ __forceinline__ v16bf load_b_frag(const bf16* base, int stride,
                                                    int lane) {
  int nr = lane & 15, kh = lane >> 4;
  const bf16* p = base + nr * stride + kh * 8;
  v8bf lo = *(const v8bf*)p;
  v8bf hi = *(const v8bf*)(p + 16);
  v16bf b;
#pragma unroll
  for (int i = 0; i < 8; ++i) { b[i] = lo[i]; b[i + 8] = hi[i]; }
  return b;
}

// Async global->LDS staging of one contiguous 64B slice per lane
// (gfx1250 GLOBAL_LOAD_ASYNC_TO_LDS_B128; tracked by ASYNCcnt).
static __device__ __forceinline__ void async_copy64(unsigned lds_addr,
                                                    unsigned long long gaddr) {
  asm volatile(
      "global_load_async_to_lds_b128 %0, %1, off\n\t"
      "global_load_async_to_lds_b128 %0, %1, off offset:16\n\t"
      "global_load_async_to_lds_b128 %0, %1, off offset:32\n\t"
      "global_load_async_to_lds_b128 %0, %1, off offset:48"
      :: "v"(lds_addr), "v"(gaddr)
      : "memory");
}

static __device__ __forceinline__ void wait_async0() {
  asm volatile("s_wait_asynccnt 0" ::: "memory");
}

// ---------------------------------------------------------------------------
// f32 -> bf16 conversion
// ---------------------------------------------------------------------------
__global__ void cvt_f32_bf16(const float* __restrict__ src, bf16* __restrict__ dst,
                             int n) {
  int i = blockIdx.x * blockDim.x + threadIdx.x;
  if (i < n) dst[i] = (bf16)src[i];
}

// ---------------------------------------------------------------------------
// Tiled bf16 WMMA GEMM:  C[M x NB] = A[M x K] @ Bw[NB x K]^T
// Block tile 128x128, K-step 32, 8 waves of 32 (each wave: 32x64 = 2x4 WMMA).
// mode 0: store f32 C.   mode 1: fused q/k/v silu (bf16), f logsigmoid, g1 raw.
// ---------------------------------------------------------------------------
__global__ __launch_bounds__(256) void gemm_bf16(
    const bf16* __restrict__ A, const bf16* __restrict__ Bw, int K, int NB,
    int mode, float* __restrict__ Cout, int ldc, bf16* __restrict__ qb,
    bf16* __restrict__ kb, bf16* __restrict__ vb, float* __restrict__ fo,
    float* __restrict__ g1o) {
  __shared__ bf16 lA[128 * 40];  // stride 40 (80B) keeps 16B alignment
  __shared__ bf16 lB[128 * 40];

  const int tid = threadIdx.x;
  const int lane = tid & 31, w = tid >> 5;
  const int m0 = blockIdx.y * 128, n0 = blockIdx.x * 128;
  const int lr = tid >> 1, half = tid & 1;  // cooperative loads: 2 thr / row
  const int wm = (w & 3) * 32, wn = (w >> 2) * 64;

  v8f zero = {};
  v8f acc[2][4];
#pragma unroll
  for (int i = 0; i < 2; ++i)
#pragma unroll
    for (int j = 0; j < 4; ++j) acc[i][j] = zero;

  for (int kk = 0; kk < K; kk += 32) {
    // ---- stage A tile (128x32 bf16) ----
    {
      const uint4* gp = (const uint4*)(A + (size_t)(m0 + lr) * K + kk + half * 16);
      uint4 a0 = gp[0], a1 = gp[1];
      *(uint4*)&lA[lr * 40 + half * 16 + 0] = a0;
      *(uint4*)&lA[lr * 40 + half * 16 + 8] = a1;
      if (kk + 32 < K)
        __builtin_prefetch(A + (size_t)(m0 + lr) * K + kk + 32, 0, 1);
    }
    // ---- stage B tile (128 rows of Bw, n-major) ----
    {
      int rb = n0 + lr;
      uint4 b0 = {0, 0, 0, 0}, b1 = {0, 0, 0, 0};
      if (rb < NB) {
        const uint4* gp = (const uint4*)(Bw + (size_t)rb * K + kk + half * 16);
        b0 = gp[0];
        b1 = gp[1];
      }
      *(uint4*)&lB[lr * 40 + half * 16 + 0] = b0;
      *(uint4*)&lB[lr * 40 + half * 16 + 8] = b1;
    }
    __syncthreads();
#pragma unroll
    for (int i = 0; i < 2; ++i) {
      v16bf af = load_a_frag(&lA[(wm + 16 * i) * 40], 40, lane);
#pragma unroll
      for (int j = 0; j < 4; ++j) {
        v16bf bf_ = load_b_frag(&lB[(wn + 16 * j) * 40], 40, lane);
        acc[i][j] = wmma_bf(af, bf_, acc[i][j]);
      }
    }
    __syncthreads();
  }

  // ---- epilogue ----
#pragma unroll
  for (int i = 0; i < 2; ++i)
#pragma unroll
    for (int j = 0; j < 4; ++j)
#pragma unroll
      for (int r = 0; r < 8; ++r) {
        int m = m0 + wm + 16 * i + ((lane >> 4) << 3) + r;
        int n = n0 + wn + 16 * j + (lane & 15);
        float v = acc[i][j][r];
        if (mode == 0) {
          Cout[(size_t)m * ldc + n] = v;
        } else {
          if (n < 3 * D_) {  // q / k / v : silu, store bf16
            float sv = v / (1.f + __expf(-v));
            int seg = n >> 10, col = n & 1023;
            bf16* dst = (seg == 0) ? qb : (seg == 1) ? kb : vb;
            dst[(size_t)m * D_ + col] = (bf16)sv;
          } else if (n < 3 * D_ + H_) {  // f : logsigmoid, f32
            fo[m * H_ + (n - 3 * D_)] = -log1pf(__expf(-v));
          } else if (n < NC_) {  // g1 : raw f32
            g1o[m * HD_ + (n - 3 * D_ - H_)] = v;
          }
        }
      }
}

// ---------------------------------------------------------------------------
// L2-normalize k per head row (64 elems, contiguous since col = h*64+d)
// ---------------------------------------------------------------------------
__global__ void knorm_kernel(bf16* __restrict__ kb, int rows) {
  int r = blockIdx.x * blockDim.x + threadIdx.x;
  if (r >= rows) return;
  bf16* p = kb + (size_t)r * HD_;
  float ss = 0.f;
  for (int i = 0; i < HD_; ++i) {
    float v = (float)p[i];
    ss += v * v;
  }
  float sc = 1.f / fmaxf(sqrtf(ss), 1e-12f);
  for (int i = 0; i < HD_; ++i) p[i] = (bf16)((float)p[i] * sc);
}

// ---------------------------------------------------------------------------
// Chunked GLA scan. One block per (b,h); 4 waves; chunk C=64; state 64x64 f32.
// Per chunk (WMMA, K-dim 64 = 2 steps):
//   O_inter = (Q * e^{F_t}) @ S_prev
//   P       = mask_causal( Qs @ (K * e^{-F_s})^T )
//   O      += P @ V
//   S_new   = e^{F63} * S_prev + (K * e^{F63-F_s})^T @ V
// State kept as T = S^T (f32, n-major for B-fragments of O_inter).
// Q/K tiles staged via async global->LDS (ASYNCcnt); V transposed manually.
// ---------------------------------------------------------------------------
__global__ __launch_bounds__(128) void gla_scan(const bf16* __restrict__ qb,
                                                const bf16* __restrict__ kb,
                                                const bf16* __restrict__ vb,
                                                const float* __restrict__ fdec,
                                                float* __restrict__ o) {
  __shared__ bf16 tQ[64 * 72];   // Q, then Qs (row-major [t][d])
  __shared__ bf16 tK[64 * 72];   // K, then Kinv (row-major [s][d])
  __shared__ bf16 tKT[64 * 72];  // Kdec transposed [d][s]
  __shared__ bf16 tVT[64 * 72];  // V transposed   [d][s]
  __shared__ bf16 tPS[64 * 72];  // phase 1: bf16 copy of T; phase 2: masked P
  __shared__ float tT[64 * 68];  // state S^T (f32)
  __shared__ float Fc[64];       // in-chunk cumulative log-decay

  const int tid = threadIdx.x;
  const int lane = tid & 31, w = tid >> 5;
  const int bh = blockIdx.x;
  const int b = bh >> 4, h = bh & 15;
  const size_t base_tok = (size_t)b * N_;
  const int srow = tid >> 1, half = tid & 1;  // loading: 2 threads per row
  const int mrow = (lane >> 4) * 8, ncol = lane & 15;

  const unsigned ldsQ = (unsigned)(size_t)&tQ[srow * 72 + half * 32];
  const unsigned ldsK = (unsigned)(size_t)&tK[srow * 72 + half * 32];

  for (int i = tid; i < 64 * 68; i += 128) tT[i] = 0.f;

  for (int ci = 0; ci < N_ / 64; ++ci) {
    const int c0 = ci * 64;
    __syncthreads();  // previous chunk fully retired; tiles free

    // ---- load Q,K async (row-major), V (transposed), f, Sb = bf16(T) ----
    {
      size_t g = (base_tok + c0 + srow) * (size_t)D_ + h * HD_ + half * 32;
      async_copy64(ldsQ, (unsigned long long)(qb + g));
      async_copy64(ldsK, (unsigned long long)(kb + g));
      const uint4* vg = (const uint4*)(vb + g);
      uint4 vv[4];
#pragma unroll
      for (int i = 0; i < 4; ++i) vv[i] = vg[i];
      const bf16* vh = reinterpret_cast<const bf16*>(vv);
#pragma unroll
      for (int e = 0; e < 32; ++e) tVT[(half * 32 + e) * 72 + srow] = vh[e];
      if (tid < 64) Fc[tid] = fdec[(base_tok + c0 + tid) * H_ + h];
      for (int i = tid; i < 64 * 64; i += 128) {
        int j = i >> 6, ii = i & 63;
        tPS[j * 72 + ii] = (bf16)tT[j * 68 + ii];
      }
      wait_async0();  // Q/K tiles landed in LDS
    }
    __syncthreads();
    if (tid == 0) {  // inclusive prefix of log-decays (serial, tiny)
      float a = 0.f;
      for (int i = 0; i < 64; ++i) { a += Fc[i]; Fc[i] = a; }
    }
    __syncthreads();
    const float F63 = Fc[63];

    // ---- scale: Qs = Q*e^{F_t}; Kinv = K*e^{-F_s}; KdecT = (K*e^{F63-F_s})^T
    {
      float fq = __expf(Fc[srow]);
      float fki = __expf(-Fc[srow]);
      float fkd = __expf(F63 - Fc[srow]);
#pragma unroll
      for (int e = 0; e < 32; ++e) {
        int d = half * 32 + e;
        tQ[srow * 72 + d] = (bf16)((float)tQ[srow * 72 + d] * fq);
        float kvv = (float)tK[srow * 72 + d];
        tK[srow * 72 + d] = (bf16)(kvv * fki);
        tKT[d * 72 + srow] = (bf16)(kvv * fkd);
      }
    }
    __syncthreads();

    v16bf af0 = load_a_frag(&tQ[(16 * w) * 72 + 0], 72, lane);
    v16bf af1 = load_a_frag(&tQ[(16 * w) * 72 + 32], 72, lane);

    // ---- O_inter = Qs @ S_prev  (B-frags from tPS = rows of T) ----
    v8f oacc[4];
#pragma unroll
    for (int jt = 0; jt < 4; ++jt) {
      v8f z = {};
      z = wmma_bf(af0, load_b_frag(&tPS[(16 * jt) * 72 + 0], 72, lane), z);
      z = wmma_bf(af1, load_b_frag(&tPS[(16 * jt) * 72 + 32], 72, lane), z);
      oacc[jt] = z;
    }
    __syncthreads();  // all waves done reading Sb before P overwrites tPS

    // ---- P = Qs @ Kinv^T, causal mask, store bf16 into tPS ----
#pragma unroll
    for (int jt = 0; jt < 4; ++jt) {
      v8f z = {};
      z = wmma_bf(af0, load_b_frag(&tK[(16 * jt) * 72 + 0], 72, lane), z);
      z = wmma_bf(af1, load_b_frag(&tK[(16 * jt) * 72 + 32], 72, lane), z);
#pragma unroll
      for (int r = 0; r < 8; ++r) {
        int t = 16 * w + mrow + r;
        int sc = 16 * jt + ncol;
        tPS[t * 72 + sc] = (sc <= t) ? (bf16)z[r] : (bf16)0.f;
      }
    }
    // same-wave LDS RAW (wave wrote its own P rows): DS ops are in-order

    // ---- O += P @ V ; store O ----
    v16bf ap0 = load_a_frag(&tPS[(16 * w) * 72 + 0], 72, lane);
    v16bf ap1 = load_a_frag(&tPS[(16 * w) * 72 + 32], 72, lane);
#pragma unroll
    for (int jt = 0; jt < 4; ++jt) {
      oacc[jt] = wmma_bf(ap0, load_b_frag(&tVT[(16 * jt) * 72 + 0], 72, lane), oacc[jt]);
      oacc[jt] = wmma_bf(ap1, load_b_frag(&tVT[(16 * jt) * 72 + 32], 72, lane), oacc[jt]);
#pragma unroll
      for (int r = 0; r < 8; ++r) {
        int t = 16 * w + mrow + r;
        int d = 16 * jt + ncol;
        o[(base_tok + c0 + t) * (size_t)D_ + h * HD_ + d] = oacc[jt][r];
      }
    }

    // ---- state: T[n][m] = e^{F63} T[n][m] + (Kdec^T @ V)[m][n] ----
    v16bf ak0 = load_a_frag(&tKT[(16 * w) * 72 + 0], 72, lane);
    v16bf ak1 = load_a_frag(&tKT[(16 * w) * 72 + 32], 72, lane);
    const float lam = __expf(F63);
#pragma unroll
    for (int jt = 0; jt < 4; ++jt) {
      int n = 16 * jt + ncol;
      v8f sacc;
#pragma unroll
      for (int r = 0; r < 8; ++r) sacc[r] = lam * tT[n * 68 + 16 * w + mrow + r];
      sacc = wmma_bf(ak0, load_b_frag(&tVT[(16 * jt) * 72 + 0], 72, lane), sacc);
      sacc = wmma_bf(ak1, load_b_frag(&tVT[(16 * jt) * 72 + 32], 72, lane), sacc);
#pragma unroll
      for (int r = 0; r < 8; ++r) tT[n * 68 + 16 * w + mrow + r] = sacc[r];
    }
    // loop-top barrier orders tT writes vs next chunk's reads
  }
}

// ---------------------------------------------------------------------------
// gate = sigmoid(g1 @ Wg2^T); o *= gate; LayerNorm(no bias) * ln_w; store bf16
// One block per token row (1024 cols, 256 threads x 4).
// ---------------------------------------------------------------------------
__global__ __launch_bounds__(256) void gate_ln(const float* __restrict__ o,
                                               const float* __restrict__ g1,
                                               const float* __restrict__ Wg2,
                                               const float* __restrict__ lnw,
                                               bf16* __restrict__ oln) {
  __shared__ float g1s[HD_];
  __shared__ float rs[256], rq[256];
  const int m = blockIdx.x, tid = threadIdx.x;
  if (tid < HD_) g1s[tid] = g1[(size_t)m * HD_ + tid];
  __syncthreads();
  float gv[4];
  float sum = 0.f, sq = 0.f;
#pragma unroll
  for (int p = 0; p < 4; ++p) {
    int c = tid + 256 * p;
    const float* wr = Wg2 + (size_t)c * HD_;
    float z = 0.f;
    for (int i = 0; i < HD_; ++i) z += g1s[i] * wr[i];
    float gate = 1.f / (1.f + __expf(-z));
    float val = o[(size_t)m * D_ + c] * gate;
    gv[p] = val;
    sum += val;
    sq += val * val;
  }
  rs[tid] = sum;
  rq[tid] = sq;
  __syncthreads();
  for (int st = 128; st > 0; st >>= 1) {
    if (tid < st) { rs[tid] += rs[tid + st]; rq[tid] += rq[tid + st]; }
    __syncthreads();
  }
  float mu = rs[0] * (1.f / (float)D_);
  float var = rq[0] * (1.f / (float)D_) - mu * mu;
  float inv = rsqrtf(var + 1e-5f);
#pragma unroll
  for (int p = 0; p < 4; ++p) {
    int c = tid + 256 * p;
    oln[(size_t)m * D_ + c] = (bf16)((gv[p] - mu) * inv * lnw[c]);
  }
}

// ---------------------------------------------------------------------------
// Host orchestration
// ---------------------------------------------------------------------------
extern "C" void kernel_launch(void* const* d_in, const int* in_sizes, int n_in,
                              void* d_out, int out_size, void* d_ws,
                              size_t ws_size, hipStream_t stream) {
  (void)in_sizes; (void)n_in; (void)out_size; (void)ws_size;
  const float* x   = (const float*)d_in[0];
  const float* Wq  = (const float*)d_in[1];
  const float* Wk  = (const float*)d_in[2];
  const float* Wv  = (const float*)d_in[3];
  const float* Wf  = (const float*)d_in[4];
  const float* Wg1 = (const float*)d_in[5];
  const float* Wg2 = (const float*)d_in[6];
  const float* lnw = (const float*)d_in[7];
  const float* Wo  = (const float*)d_in[8];

  char* ws = (char*)d_ws;
  size_t off = 0;
  auto take = [&](size_t bytes) {
    size_t cur = off;
    off += (bytes + 255) & ~(size_t)255;
    return cur;
  };
  bf16*  xb   = (bf16*) (ws + take((size_t)M_ * D_ * 2));
  bf16*  wcat = (bf16*) (ws + take((size_t)NC_ * D_ * 2));
  bf16*  wob  = (bf16*) (ws + take((size_t)D_ * D_ * 2));
  bf16*  qb   = (bf16*) (ws + take((size_t)M_ * D_ * 2));
  bf16*  kb   = (bf16*) (ws + take((size_t)M_ * D_ * 2));
  bf16*  vbuf = (bf16*) (ws + take((size_t)M_ * D_ * 2));
  float* fo   = (float*)(ws + take((size_t)M_ * H_ * 4));
  float* g1b  = (float*)(ws + take((size_t)M_ * HD_ * 4));
  float* obuf = (float*)(ws + take((size_t)M_ * D_ * 4));
  bf16*  oln  = (bf16*) (ws + take((size_t)M_ * D_ * 2));

  auto cv = [&](const float* s, bf16* d, int n) {
    cvt_f32_bf16<<<(n + 255) / 256, 256, 0, stream>>>(s, d, n);
  };
  cv(x, xb, M_ * D_);
  cv(Wq, wcat, D_ * D_);
  cv(Wk, wcat + (size_t)D_ * D_, D_ * D_);
  cv(Wv, wcat + (size_t)2 * D_ * D_, D_ * D_);
  cv(Wf, wcat + (size_t)3 * D_ * D_, H_ * D_);
  cv(Wg1, wcat + (size_t)3 * D_ * D_ + H_ * D_, HD_ * D_);
  cv(Wo, wob, D_ * D_);

  // fused qkv/f/g1 projection (16384 x 3152 x 1024)
  dim3 grid1((NC_ + 127) / 128, M_ / 128);
  gemm_bf16<<<grid1, 256, 0, stream>>>(xb, wcat, D_, NC_, 1, nullptr, 0, qb, kb,
                                       vbuf, fo, g1b);
  // normalize k per head
  knorm_kernel<<<(M_ * H_) / 256, 256, 0, stream>>>(kb, M_ * H_);
  // chunked GLA scan: one block per (b,h)
  gla_scan<<<B_ * H_, 128, 0, stream>>>(qb, kb, vbuf, fo, obuf);
  // output gate + layernorm
  gate_ln<<<M_, 256, 0, stream>>>(obuf, g1b, Wg2, lnw, oln);
  // final projection (16384 x 1024 x 1024) -> f32 output
  dim3 grid2(D_ / 128, M_ / 128);
  gemm_bf16<<<grid2, 256, 0, stream>>>(oln, wob, D_, D_, 0, (float*)d_out, D_,
                                       nullptr, nullptr, nullptr, nullptr,
                                       nullptr);
}
